// LatentBlock_45114336477545
// MI455X (gfx1250) — compile-verified, matrix operand order checked by reading, not run on hardware
//
#include <hip/hip_runtime.h>
#include <math.h>

// ---------------- problem dims ----------------
#define D_    1024
#define H_    16
#define DH_   64
#define E_    8
#define TOPK_ 2
#define DFF_  4096
#define B_    4
#define L_    512
#define T_    2048
#define NL_   (B_ * L_)        // 2048 latent rows
#define NT_   (B_ * T_)        // 8192 token rows
#define NASSIGN_ (NL_ * TOPK_) // 4096 expert assignments

// ---------------- types ----------------
typedef __attribute__((ext_vector_type(16))) __bf16 v16bf;
typedef __attribute__((ext_vector_type(8)))  float  v8f;
typedef __attribute__((ext_vector_type(4)))  unsigned int v4u;

union frag_u { v16bf v; v4u q[2]; unsigned short e[16]; };

#define WMMA_BF16(a, b, c) \
  __builtin_amdgcn_wmma_f32_16x16x32_bf16(false, (a), false, (b), (short)0, (c), false, false)

#ifndef __has_builtin
#define __has_builtin(x) 0
#endif
#if __has_builtin(__builtin_amdgcn_tensor_load_to_lds) && \
    __has_builtin(__builtin_amdgcn_s_wait_tensorcnt)
#define HAVE_TDM 1
#else
#define HAVE_TDM 0
#endif

__device__ __forceinline__ unsigned short f2bf(float f) {
  union { float f; unsigned u; } x; x.f = f;
  unsigned r = x.u + 0x7FFFu + ((x.u >> 16) & 1u);
  return (unsigned short)(r >> 16);
}
__device__ __forceinline__ float bf2f(unsigned short h) {
  union { unsigned u; float f; } x; x.u = ((unsigned)h) << 16;
  return x.f;
}

// Load a 16x32 bf16 fragment row-slice for one lane. p points at
// element [row][k0 + (lane>=16 ? 8 : 0)] of a row-major bf16 matrix.
// elems 0..7 <- p[0..7], elems 8..15 <- p[16..23]  (ISA 16-bit A/B layout).
__device__ __forceinline__ v16bf load_frag(const unsigned short* p) {
  frag_u u;
  u.q[0] = *(const v4u*)(p);
  u.q[1] = *(const v4u*)(p + 16);
  return u.v;
}

__device__ __forceinline__ v8f zero8() {
  v8f z = {0.f, 0.f, 0.f, 0.f, 0.f, 0.f, 0.f, 0.f};
  return z;
}

// ---------------- TDM helpers ----------------
// LDS byte offset of a generic pointer into shared memory: per ISA 10.2 the
// low 32 bits of a shared-aperture generic address ARE the LDS byte address.
__device__ __forceinline__ unsigned lds_raw_off(const void* p) {
  return (unsigned)(unsigned long long)(size_t)p;
}

#define LDSROW 40              // padded LDS row stride in bf16 elems (64B data + 16B pad)
#define SLAB   (128 * LDSROW)  // one 128x32 tile slab (elems)

#if HAVE_TDM
typedef __attribute__((ext_vector_type(4))) unsigned int u32x4_t;
typedef __attribute__((ext_vector_type(8))) int i32x8_t;
typedef __attribute__((ext_vector_type(4))) int i32x4_t;

// DMA a 128-row x 32-col bf16 tile (row stride ldElems) from global to LDS,
// with 16B padding after each 64B row => LDS row stride 80B.
// D# layout per CDNA5 ISA 8.3-8.5. 6-arg builtin form (clang-23 / therock).
__device__ __forceinline__ void tdm_load_tile(const unsigned short* gsrc,
                                              unsigned ldsOff, int ldElems) {
  unsigned long long ga = (unsigned long long)(size_t)gsrc;
  u32x4_t g0;
  g0[0] = 1u;                                                  // count=1, user D#
  g0[1] = ldsOff;                                              // lds_addr
  g0[2] = (unsigned)(ga & 0xFFFFFFFFull);                      // global_addr[31:0]
  g0[3] = (unsigned)((ga >> 32) & 0x1FFFFFFull) | (2u << 30);  // [56:32] | type=2
  i32x8_t g1;
  // data_size=2B (code1), pad_enable, pad_interval=16DW(code3), pad_amount=4DW(code3)
  g1[0] = (1 << 16) | (1 << 20) | (3 << 22) | (3 << 25);
  unsigned td0 = (unsigned)ldElems;   // tensor_dim0 (>= tile extent; no OOB by construction)
  unsigned td1 = 1u << 20;            // tensor_dim1 (large)
  g1[1] = (int)((td0 & 0xFFFFu) << 16);                  // atomic_barrier_addr=0 | dim0 lo
  g1[2] = (int)((td0 >> 16) | ((td1 & 0xFFFFu) << 16));  // dim0 hi | dim1 lo
  g1[3] = (int)((td1 >> 16) | (32u << 16));              // dim1 hi | tile_dim0=32
  g1[4] = 128;                                           // tile_dim1=128, tile_dim2=0
  g1[5] = ldElems;                                       // tensor_dim0_stride lo32
  g1[6] = 0;                                             // stride0 hi | stride1 lo
  g1[7] = 0;
  i32x4_t gz4 = {0, 0, 0, 0};
  i32x8_t gz8 = {0, 0, 0, 0, 0, 0, 0, 0};
  __builtin_amdgcn_tensor_load_to_lds(g0, g1, gz4, gz4, gz8, 0);
}
#endif

// ---------------- fp32 -> bf16 convert ----------------
__global__ void cvt_kernel(const float* __restrict__ src, unsigned short* __restrict__ dst, long n) {
  long i = (long)blockIdx.x * blockDim.x + threadIdx.x;
  if (i < n) dst[i] = f2bf(src[i]);
}

// ---------------- LayerNorm -> bf16 ----------------
__global__ __launch_bounds__(256) void ln_kernel(const float* __restrict__ x,
                                                 const float* __restrict__ g,
                                                 const float* __restrict__ b,
                                                 unsigned short* __restrict__ out) {
  int row = blockIdx.x;
  const float* xr = x + (long)row * D_;
  float s = 0.f, s2 = 0.f;
  for (int i = threadIdx.x; i < D_; i += 256) { float v = xr[i]; s += v; s2 += v * v; }
  for (int m = 16; m >= 1; m >>= 1) { s += __shfl_xor(s, m, 32); s2 += __shfl_xor(s2, m, 32); }
  __shared__ float red[2][8];
  int wid = threadIdx.x >> 5;
  if ((threadIdx.x & 31) == 0) { red[0][wid] = s; red[1][wid] = s2; }
  __syncthreads();
  float ts = 0.f, ts2 = 0.f;
#pragma unroll
  for (int w = 0; w < 8; ++w) { ts += red[0][w]; ts2 += red[1][w]; }
  float mean = ts / D_;
  float var = ts2 / D_ - mean * mean;
  float inv = rsqrtf(var + 1e-5f);
  unsigned short* orow = out + (long)row * D_;
  for (int i = threadIdx.x; i < D_; i += 256)
    orow[i] = f2bf((xr[i] - mean) * inv * g[i] + b[i]);
}

// ---------------- dense WMMA GEMM with TDM double-buffered LDS staging -------
// C[M,N] = act(A[M,K] * W[N,K]^T + bias) (+ residual). 128x128 block tile,
// 8 waves (2m x 4n), each wave 64x32. A/B k-slices (128x32 bf16) DMA'd to LDS
// by the Tensor Data Mover, double buffered on TENSORcnt.
// flags: 1=GELU, 2=bf16 output.
__global__ __launch_bounds__(256) void gemm_tdm_kernel(
    const unsigned short* __restrict__ A, int lda,
    const unsigned short* __restrict__ W, int ldw,
    const float* __restrict__ bias,
    const float* __restrict__ residual,
    void* __restrict__ Cout, int ldc,
    int K, int flags) {
  __shared__ unsigned short smem[2][2 * SLAB];   // [buf][A slab | B slab]
  const int lane = threadIdx.x & 31;
  const int wv = threadIdx.x >> 5;
  const int wm = wv & 1, wn = wv >> 1;
  const int l15 = lane & 15, lhi = lane >> 4;
  const int m0 = blockIdx.y * 128, n0 = blockIdx.x * 128;

  const unsigned short* gA = A + (size_t)m0 * lda;
  const unsigned short* gW = W + (size_t)n0 * ldw;

  v8f acc[4][2];
#pragma unroll
  for (int i = 0; i < 4; ++i)
#pragma unroll
    for (int j = 0; j < 2; ++j) acc[i][j] = zero8();

  const int nSl = K >> 5;
#if HAVE_TDM
  const unsigned ldsA0 = lds_raw_off(&smem[0][0]);
  const unsigned ldsA1 = lds_raw_off(&smem[1][0]);
  const unsigned ldsB0 = lds_raw_off(&smem[0][SLAB]);
  const unsigned ldsB1 = lds_raw_off(&smem[1][SLAB]);
  if (wv == 0) {
    tdm_load_tile(gA, ldsA0, lda);
    tdm_load_tile(gW, ldsB0, ldw);
  }
#endif
  for (int kc = 0; kc < nSl; ++kc) {
#if HAVE_TDM
    const int cur = kc & 1;
    if (wv == 0) {
      if (kc + 1 < nSl) {
        tdm_load_tile(gA + (kc + 1) * 32, cur ? ldsA0 : ldsA1, lda);
        tdm_load_tile(gW + (kc + 1) * 32, cur ? ldsB0 : ldsB1, ldw);
        __builtin_amdgcn_s_wait_tensorcnt((short)2);  // current slice done, next in flight
      } else {
        __builtin_amdgcn_s_wait_tensorcnt((short)0);
      }
    }
    __syncthreads();
#else
    const int cur = 0;
    __syncthreads();
    for (int c = threadIdx.x; c < 1024; c += 256) {   // synchronous staging fallback
      int isB = c >> 9, cc = c & 511;
      int row = cc >> 2, c4 = cc & 3;
      const unsigned short* src =
          (isB ? gW : gA) + (size_t)row * (isB ? ldw : lda) + kc * 32 + c4 * 8;
      *(v4u*)&smem[0][isB * SLAB + row * LDSROW + c4 * 8] = *(const v4u*)src;
    }
    __syncthreads();
#endif
    const unsigned short* Ab = &smem[cur][0];
    const unsigned short* Bb = &smem[cur][SLAB];
    v16bf b0 = load_frag(&Bb[(wn * 32 + l15) * LDSROW + (lhi << 3)]);
    v16bf b1 = load_frag(&Bb[(wn * 32 + 16 + l15) * LDSROW + (lhi << 3)]);
#pragma unroll
    for (int i = 0; i < 4; ++i) {
      v16bf a = load_frag(&Ab[(wm * 64 + i * 16 + l15) * LDSROW + (lhi << 3)]);
      acc[i][0] = WMMA_BF16(a, b0, acc[i][0]);
      acc[i][1] = WMMA_BF16(a, b1, acc[i][1]);
    }
    __syncthreads();
  }

  const bool gelu = (flags & 1) != 0, obf = (flags & 2) != 0;
#pragma unroll
  for (int i = 0; i < 4; ++i)
#pragma unroll
    for (int j = 0; j < 2; ++j) {
      int n = n0 + wn * 32 + j * 16 + l15;
      float bv = bias ? bias[n] : 0.f;
#pragma unroll
      for (int r = 0; r < 8; ++r) {
        int m = m0 + wm * 64 + i * 16 + r + (lhi << 3);
        float v = acc[i][j][r] + bv;
        if (gelu) v = 0.5f * v * (1.f + erff(v * 0.70710678118654752f));
        long oi = (long)m * ldc + n;
        if (residual) v += residual[oi];
        if (obf) ((unsigned short*)Cout)[oi] = f2bf(v);
        else     ((float*)Cout)[oi] = v;
      }
    }
}

// ---------------- grouped WMMA GEMM (MoE experts, gather-indexed A) ----------
// flags: 1=GELU, 2=bf16 output, 4=grouped, 8=indirect A rows through rowTok.
__global__ __launch_bounds__(256) void gemm_kernel(
    const unsigned short* __restrict__ A, int lda,
    const unsigned short* __restrict__ W, long wStrideE, int ldw,
    const float* __restrict__ bias, int biasStrideE,
    const float* __restrict__ residual,
    void* __restrict__ Cout, int ldc,
    int M, int K, int flags,
    const int* __restrict__ gCounts, const int* __restrict__ gOffsets,
    const int* __restrict__ rowTok) {
  const int lane = threadIdx.x & 31;
  const int wv = threadIdx.x >> 5;
  const int wm = wv & 1, wn = wv >> 1;
  const int l15 = lane & 15, lhi = lane >> 4;

  const bool grouped = (flags & 4) != 0;
  int cnt = M, outRow0 = 0;
  if (grouped) {
    int e = blockIdx.z;
    cnt = gCounts[e];
    if ((int)blockIdx.y * 128 >= cnt) return;
    outRow0 = gOffsets[e];
    W += (long)e * wStrideE;
    bias += (long)e * biasStrideE;
  }
  const int m0 = blockIdx.y * 128 + wm * 64;
  const int n0 = blockIdx.x * 128 + wn * 32;

  const unsigned short* aP[4];
#pragma unroll
  for (int i = 0; i < 4; ++i) {
    int m = m0 + i * 16 + l15;
    int mc = m < cnt ? m : cnt - 1;
    long arow;
    if (grouped) {
      long gr = (long)outRow0 + mc;
      arow = (flags & 8) ? (long)rowTok[gr] : gr;
    } else {
      arow = mc;
    }
    aP[i] = A + arow * (long)lda + (lhi << 3);
  }
  const unsigned short* bP[2];
#pragma unroll
  for (int j = 0; j < 2; ++j)
    bP[j] = W + (long)(n0 + j * 16 + l15) * ldw + (lhi << 3);

  v8f acc[4][2];
#pragma unroll
  for (int i = 0; i < 4; ++i)
#pragma unroll
    for (int j = 0; j < 2; ++j) acc[i][j] = zero8();

  for (int k0 = 0; k0 < K; k0 += 32) {
    v16bf bfr0 = load_frag(bP[0] + k0);
    v16bf bfr1 = load_frag(bP[1] + k0);
#pragma unroll
    for (int i = 0; i < 4; ++i) {
      v16bf afr = load_frag(aP[i] + k0);
      acc[i][0] = WMMA_BF16(afr, bfr0, acc[i][0]);
      acc[i][1] = WMMA_BF16(afr, bfr1, acc[i][1]);
    }
  }

  const bool gelu = (flags & 1) != 0, obf = (flags & 2) != 0;
#pragma unroll
  for (int i = 0; i < 4; ++i)
#pragma unroll
    for (int j = 0; j < 2; ++j) {
      int n = n0 + j * 16 + l15;
      float bv = bias ? bias[n] : 0.f;
#pragma unroll
      for (int r = 0; r < 8; ++r) {
        int m = m0 + i * 16 + r + (lhi << 3);
        if (grouped && m >= cnt) continue;
        float v = acc[i][j][r] + bv;
        if (gelu) v = 0.5f * v * (1.f + erff(v * 0.70710678118654752f));
        long orow = grouped ? (long)outRow0 + m : (long)m;
        long oi = orow * (long)ldc + n;
        if (residual) v += residual[oi];
        if (obf) ((unsigned short*)Cout)[oi] = f2bf(v);
        else     ((float*)Cout)[oi] = v;
      }
    }
}

// ---------------- flash attention (one wave per 16 queries) ----------------
__global__ __launch_bounds__(32) void attn_kernel(
    const unsigned short* __restrict__ Qm, int ldq, int qc0,
    const unsigned short* __restrict__ Km, int ldk, int kc0,
    const unsigned short* __restrict__ Vm, int ldv, int vc0,
    unsigned short* __restrict__ Om,
    int Lq, int Tk, float scale) {
  const int lane = threadIdx.x;
  const int l15 = lane & 15, lhi = lane >> 4;
  const int qt = blockIdx.x, h = blockIdx.y, b = blockIdx.z;
  const long qrow0 = (long)b * Lq + qt * 16;
  const long krow0 = (long)b * Tk;
  const int qcol = qc0 + h * DH_;
  const int kcol = kc0 + h * DH_;
  const int vcol0 = vc0 + h * DH_;

  v16bf qf[2];
#pragma unroll
  for (int c = 0; c < 2; ++c)
    qf[c] = load_frag(Qm + (size_t)(qrow0 + l15) * ldq + qcol + c * 32 + (lhi << 3));

  v8f o[4];
#pragma unroll
  for (int j = 0; j < 4; ++j) o[j] = zero8();
  float mrow[8], lrow[8];
#pragma unroll
  for (int r = 0; r < 8; ++r) { mrow[r] = -1e30f; lrow[r] = 0.f; }

  __shared__ unsigned short Pl[16 * 32];

  for (int t0 = 0; t0 < Tk; t0 += 32) {
    v8f s0 = zero8(), s1 = zero8();
#pragma unroll
    for (int c = 0; c < 2; ++c) {
      const unsigned short* kp0 = Km + (size_t)(krow0 + t0 + l15) * ldk + kcol + c * 32 + (lhi << 3);
      const unsigned short* kp1 = Km + (size_t)(krow0 + t0 + 16 + l15) * ldk + kcol + c * 32 + (lhi << 3);
      s0 = WMMA_BF16(qf[c], load_frag(kp0), s0);
      s1 = WMMA_BF16(qf[c], load_frag(kp1), s1);
    }
    s0 = s0 * scale;
    s1 = s1 * scale;

#pragma unroll
    for (int r = 0; r < 8; ++r) {
      float t = fmaxf(s0[r], s1[r]);
#pragma unroll
      for (int mk = 1; mk < 16; mk <<= 1) t = fmaxf(t, __shfl_xor(t, mk, 16));
      float nm = fmaxf(mrow[r], t);
      float al = __expf(mrow[r] - nm);
      float p0 = __expf(s0[r] - nm);
      float p1 = __expf(s1[r] - nm);
      float rs = p0 + p1;
#pragma unroll
      for (int mk = 1; mk < 16; mk <<= 1) rs += __shfl_xor(rs, mk, 16);
      lrow[r] = lrow[r] * al + rs;
      mrow[r] = nm;
      s0[r] = p0; s1[r] = p1;
      o[0][r] *= al; o[1][r] *= al; o[2][r] *= al; o[3][r] *= al;
    }

    __syncthreads();
#pragma unroll
    for (int r = 0; r < 8; ++r) {
      int m = r + (lhi << 3);
      Pl[m * 32 + l15]      = f2bf(s0[r]);
      Pl[m * 32 + 16 + l15] = f2bf(s1[r]);
    }
    __syncthreads();
    frag_u pu;
    pu.q[0] = *(const v4u*)&Pl[l15 * 32 + (lhi << 3)];
    pu.q[1] = *(const v4u*)&Pl[l15 * 32 + (lhi << 3) + 16];

#pragma unroll
    for (int j = 0; j < 4; ++j) {
      frag_u vu;
      const unsigned short* vcol = Vm + vcol0 + j * 16 + l15;
#pragma unroll
      for (int kk = 0; kk < 8; ++kk) {
        vu.e[kk]     = vcol[(size_t)(krow0 + t0 + (lhi << 3) + kk) * ldv];
        vu.e[kk + 8] = vcol[(size_t)(krow0 + t0 + 16 + (lhi << 3) + kk) * ldv];
      }
      o[j] = WMMA_BF16(pu.v, vu.v, o[j]);
    }
  }

#pragma unroll
  for (int j = 0; j < 4; ++j)
#pragma unroll
    for (int r = 0; r < 8; ++r) {
      int m = r + (lhi << 3);
      float val = o[j][r] / lrow[r];
      Om[(size_t)(qrow0 + m) * (H_ * DH_) + h * DH_ + j * 16 + l15] = f2bf(val);
    }
}

// ---------------- MoE routing ----------------
__global__ void zero_kernel(int* p, int n) {
  int i = blockIdx.x * blockDim.x + threadIdx.x;
  if (i < n) p[i] = 0;
}

__global__ __launch_bounds__(256) void router_kernel(
    const unsigned short* __restrict__ h, const float* __restrict__ rw,
    const float* __restrict__ rb, int* __restrict__ topIdx,
    float* __restrict__ gates, int* __restrict__ counts) {
  int wid = (int)(blockIdx.x * 8 + (threadIdx.x >> 5));
  int lane = threadIdx.x & 31;
  if (wid >= NL_) return;
  const unsigned short* hr = h + (long)wid * D_;
  float logit[E_];
#pragma unroll
  for (int e = 0; e < E_; ++e) {
    float s = 0.f;
    for (int d = lane; d < D_; d += 32) s += bf2f(hr[d]) * rw[e * D_ + d];
    for (int m = 16; m >= 1; m >>= 1) s += __shfl_xor(s, m, 32);
    logit[e] = s + rb[e];
  }
  if (lane == 0) {
    int i0 = 0; float v0 = logit[0];
#pragma unroll
    for (int e = 1; e < E_; ++e) if (logit[e] > v0) { v0 = logit[e]; i0 = e; }
    int i1 = (i0 == 0) ? 1 : 0; float v1 = logit[i1];
#pragma unroll
    for (int e = 0; e < E_; ++e)
      if (e != i0 && logit[e] > v1) { v1 = logit[e]; i1 = e; }
    float e0 = 1.f;                     // exp(v0 - v0)
    float e1 = __expf(v1 - v0);
    float den = e0 + e1;
    topIdx[wid * 2] = i0; topIdx[wid * 2 + 1] = i1;
    gates[wid * 2] = e0 / den; gates[wid * 2 + 1] = e1 / den;
    atomicAdd(&counts[i0], 1);
    atomicAdd(&counts[i1], 1);
  }
}

__global__ void scan_kernel(const int* __restrict__ counts, int* __restrict__ offsets,
                            int* __restrict__ cursor) {
  if (threadIdx.x == 0) {
    int acc = 0;
    for (int e = 0; e < E_; ++e) { offsets[e] = acc; acc += counts[e]; }
    offsets[E_] = acc;
  }
  if (threadIdx.x < E_) cursor[threadIdx.x] = 0;
}

__global__ void place_kernel(const int* __restrict__ topIdx, const int* __restrict__ offsets,
                             int* __restrict__ cursor, int* __restrict__ rowTok,
                             int* __restrict__ assignRow) {
  int n = blockIdx.x * blockDim.x + threadIdx.x;
  if (n >= NL_) return;
#pragma unroll
  for (int k = 0; k < TOPK_; ++k) {
    int e = topIdx[n * 2 + k];
    int pos = atomicAdd(&cursor[e], 1);
    int row = offsets[e] + pos;
    rowTok[row] = n;
    assignRow[n * 2 + k] = row;
  }
}

__global__ void combine_kernel(const float* __restrict__ x2, const float* __restrict__ eo,
                               const int* __restrict__ assignRow, const float* __restrict__ gates,
                               float* __restrict__ out) {
  int i = blockIdx.x * blockDim.x + threadIdx.x;
  if (i >= NL_ * D_) return;
  int n = i / D_, d = i - n * D_;
  float v = x2[i];
  v += gates[n * 2]     * eo[(long)assignRow[n * 2]     * D_ + d];
  v += gates[n * 2 + 1] * eo[(long)assignRow[n * 2 + 1] * D_ + d];
  out[i] = v;
}

// ---------------- host ----------------
extern "C" void kernel_launch(void* const* d_in, const int* in_sizes, int n_in,
                              void* d_out, int out_size, void* d_ws, size_t ws_size,
                              hipStream_t stream) {
  const float* latents  = (const float*)d_in[0];
  const float* tokens   = (const float*)d_in[1];
  const float* sa_ln_g  = (const float*)d_in[2];
  const float* sa_ln_b  = (const float*)d_in[3];
  const float* sa_in_w  = (const float*)d_in[4];
  const float* sa_in_b  = (const float*)d_in[5];
  const float* sa_out_w = (const float*)d_in[6];
  const float* sa_out_b = (const float*)d_in[7];
  const float* ca_q_ln_g  = (const float*)d_in[8];
  const float* ca_q_ln_b  = (const float*)d_in[9];
  const float* ca_kv_ln_g = (const float*)d_in[10];
  const float* ca_kv_ln_b = (const float*)d_in[11];
  const float* ca_in_w  = (const float*)d_in[12];
  const float* ca_in_b  = (const float*)d_in[13];
  const float* ca_out_w = (const float*)d_in[14];
  const float* ca_out_b = (const float*)d_in[15];
  const float* moe_ln_g = (const float*)d_in[16];
  const float* moe_ln_b = (const float*)d_in[17];
  const float* router_w = (const float*)d_in[18];
  const float* router_b = (const float*)d_in[19];
  const float* e_fc1_w  = (const float*)d_in[20];
  const float* e_fc1_b  = (const float*)d_in[21];
  const float* e_fc2_w  = (const float*)d_in[22];
  const float* e_fc2_b  = (const float*)d_in[23];
  float* out = (float*)d_out;

  char* wsp = (char*)d_ws;
  auto take = [&](size_t bytes) -> void* {
    void* p = (void*)wsp;
    wsp += (bytes + 255) & ~(size_t)255;
    return p;
  };

  unsigned short* saInW  = (unsigned short*)take((size_t)3 * D_ * D_ * 2);
  unsigned short* saOutW = (unsigned short*)take((size_t)D_ * D_ * 2);
  unsigned short* caInW  = (unsigned short*)take((size_t)3 * D_ * D_ * 2);
  unsigned short* caOutW = (unsigned short*)take((size_t)D_ * D_ * 2);
  unsigned short* fc1W   = (unsigned short*)take((size_t)E_ * DFF_ * D_ * 2);
  unsigned short* fc2W   = (unsigned short*)take((size_t)E_ * D_ * DFF_ * 2);
  unsigned short* qsa    = (unsigned short*)take((size_t)NL_ * D_ * 2);
  unsigned short* qkv    = (unsigned short*)take((size_t)NL_ * 3 * D_ * 2);
  unsigned short* attn1  = (unsigned short*)take((size_t)NL_ * D_ * 2);
  float*          x1     = (float*)take((size_t)NL_ * D_ * 4);
  unsigned short* qc     = (unsigned short*)take((size_t)NL_ * D_ * 2);
  unsigned short* kvn    = (unsigned short*)take((size_t)NT_ * D_ * 2);
  unsigned short* qproj  = (unsigned short*)take((size_t)NL_ * D_ * 2);
  unsigned short* kvproj = (unsigned short*)take((size_t)NT_ * 2 * D_ * 2);
  unsigned short* attn2  = (unsigned short*)take((size_t)NL_ * D_ * 2);
  float*          x2     = (float*)take((size_t)NL_ * D_ * 4);
  unsigned short* hmoe   = (unsigned short*)take((size_t)NL_ * D_ * 2);
  unsigned short* fc1out = (unsigned short*)take((size_t)NASSIGN_ * DFF_ * 2);
  float*          eo     = (float*)take((size_t)NASSIGN_ * D_ * 4);
  int*   topIdx    = (int*)take(NL_ * 2 * 4);
  float* gates     = (float*)take(NL_ * 2 * 4);
  int*   counts    = (int*)take(E_ * 4);
  int*   offsets   = (int*)take((E_ + 1) * 4);
  int*   cursor    = (int*)take(E_ * 4);
  int*   rowTok    = (int*)take(NASSIGN_ * 4);
  int*   assignRow = (int*)take(NASSIGN_ * 4);

  auto cvt = [&](const float* s, unsigned short* d, long n) {
    cvt_kernel<<<dim3((unsigned)((n + 255) / 256)), 256, 0, stream>>>(s, d, n);
  };
  // weights -> bf16 once per launch
  cvt(sa_in_w,  saInW,  (long)3 * D_ * D_);
  cvt(sa_out_w, saOutW, (long)D_ * D_);
  cvt(ca_in_w,  caInW,  (long)3 * D_ * D_);
  cvt(ca_out_w, caOutW, (long)D_ * D_);
  cvt(e_fc1_w,  fc1W,   (long)E_ * DFF_ * D_);
  cvt(e_fc2_w,  fc2W,   (long)E_ * D_ * DFF_);

  auto gemmDense = [&](const unsigned short* A, int lda, const unsigned short* W, int ldw,
                       const float* bias, const float* res, void* C, int ldc,
                       int M, int K, int N, int flags) {
    dim3 grid(N / 128, M / 128, 1);
    gemm_tdm_kernel<<<grid, 256, 0, stream>>>(A, lda, W, ldw, bias, res, C, ldc, K, flags);
  };
  auto gemmGrouped = [&](const unsigned short* A, int lda, const unsigned short* W,
                         long wSE, int ldw, const float* bias, int bSE, void* C, int ldc,
                         int K, int N, int flags, const int* rtok) {
    dim3 grid(N / 128, NASSIGN_ / 128, E_);
    gemm_kernel<<<grid, 256, 0, stream>>>(A, lda, W, wSE, ldw, bias, bSE, nullptr, C, ldc,
                                          0, K, flags | 4, counts, offsets, rtok);
  };

  // ---- self attention block ----
  ln_kernel<<<NL_, 256, 0, stream>>>(latents, sa_ln_g, sa_ln_b, qsa);
  gemmDense(qsa, D_, saInW, D_, sa_in_b, nullptr, qkv, 3 * D_, NL_, D_, 3 * D_, /*bf16*/2);
  attn_kernel<<<dim3(L_ / 16, H_, B_), 32, 0, stream>>>(
      qkv, 3 * D_, 0, qkv, 3 * D_, D_, qkv, 3 * D_, 2 * D_, attn1, L_, L_, 0.125f);
  gemmDense(attn1, D_, saOutW, D_, sa_out_b, latents, x1, D_, NL_, D_, D_, /*f32 out*/0);

  // ---- cross attention block ----
  ln_kernel<<<NL_, 256, 0, stream>>>(x1, ca_q_ln_g, ca_q_ln_b, qc);
  ln_kernel<<<NT_, 256, 0, stream>>>(tokens, ca_kv_ln_g, ca_kv_ln_b, kvn);
  gemmDense(qc, D_, caInW, D_, ca_in_b, nullptr, qproj, D_, NL_, D_, D_, 2);
  gemmDense(kvn, D_, caInW + (size_t)D_ * D_, D_, ca_in_b + D_, nullptr,
            kvproj, 2 * D_, NT_, D_, 2 * D_, 2);
  attn_kernel<<<dim3(L_ / 16, H_, B_), 32, 0, stream>>>(
      qproj, D_, 0, kvproj, 2 * D_, 0, kvproj, 2 * D_, D_, attn2, L_, T_, 0.125f);
  gemmDense(attn2, D_, caOutW, D_, ca_out_b, x1, x2, D_, NL_, D_, D_, 0);

  // ---- MoE ----
  ln_kernel<<<NL_, 256, 0, stream>>>(x2, moe_ln_g, moe_ln_b, hmoe);
  zero_kernel<<<1, 32, 0, stream>>>(counts, E_);
  router_kernel<<<NL_ / 8, 256, 0, stream>>>(hmoe, router_w, router_b, topIdx, gates, counts);
  scan_kernel<<<1, 32, 0, stream>>>(counts, offsets, cursor);
  place_kernel<<<NL_ / 256, 256, 0, stream>>>(topIdx, offsets, cursor, rowTok, assignRow);
  // fc1: gathered tokens -> GELU -> bf16
  gemmGrouped(hmoe, D_, fc1W, (long)DFF_ * D_, D_, e_fc1_b, DFF_,
              fc1out, DFF_, D_, DFF_, /*gelu|bf16|indirect*/ 1 | 2 | 8, rowTok);
  // fc2: assignment rows -> fp32
  gemmGrouped(fc1out, DFF_, fc2W, (long)D_ * DFF_, DFF_, e_fc2_b, D_,
              eo, D_, DFF_, D_, 0, nullptr);
  combine_kernel<<<(NL_ * D_) / 256, 256, 0, stream>>>(x2, eo, assignRow, gates, out);
}
